// RWKV7PoSTAttention_42125039239453
// MI455X (gfx1250) — matrix operand order repaired and checked by reading, not compiled, not run on hardware
//
#include <hip/hip_runtime.h>
#include <hip/hip_bf16.h>
#include <math.h>

// ---------------------------------------------------------------------------
// RWKV-7 PoST attention for MI455X (gfx1250, wave32, WMMA + TDM).
//   - bf16 is the storage format for all GEMM operands: weights converted
//     once, token-shift-mixed activations materialized as bf16, LoRA hiddens
//     and the gated output written bf16 straight from producer epilogues.
//   - GEMMs via v_wmma_f32_16x16x32_bf16: 128x128 block tile, 8 waves,
//     8 WMMAs/wave/K-step; BOTH tiles staged by the Tensor Data Mover
//     (tensor_load_to_lds, pad_enable -> conflict-free [128][34] bf16 rows,
//     OOB zero-fill on edges), double-buffered so the DMA of tile i+1
//     overlaps the WMMAs of tile i; s_wait_tensorcnt 0 drains the in-order
//     TDM queue. The inner loop has no global loads and no conversion VALU.
//   - fp32 kept for the numerically sensitive recurrence / GroupNorm path.
// ---------------------------------------------------------------------------

typedef __attribute__((ext_vector_type(16))) __bf16       v16bf;
typedef __attribute__((ext_vector_type(8)))  float        v8f;
typedef __attribute__((ext_vector_type(4)))  unsigned int v4u;
typedef __attribute__((ext_vector_type(8)))  int          v8i;
typedef __attribute__((ext_vector_type(4)))  int          v4i;

static __device__ __forceinline__ __bf16 f2bf(float f) {
  union { float f; unsigned u; } x; x.f = f;
  unsigned r = x.u + 0x7FFFu + ((x.u >> 16) & 1u);   // round-to-nearest-even
  unsigned short h = (unsigned short)(r >> 16);
  __bf16 out;
  __builtin_memcpy(&out, &h, sizeof(h));
  return out;
}

static __device__ __forceinline__ float softplus_f(float z) {
  return (z > 20.0f) ? z : log1pf(expf(z));
}

// ---------------------------------------------------------------------------
// fp32 -> bf16 bulk convert (weights).  n4 = element count / 4.
// ---------------------------------------------------------------------------
__global__ __launch_bounds__(256)
void cvt_kernel(const float* __restrict__ src, __bf16* __restrict__ dst, int n4)
{
  int i4 = blockIdx.x * 256 + threadIdx.x;
  if (i4 < n4) {
    float4 v = *(const float4*)&src[(size_t)i4 * 4];
    size_t o = (size_t)i4 * 4;
    dst[o + 0] = f2bf(v.x);
    dst[o + 1] = f2bf(v.y);
    dst[o + 2] = f2bf(v.z);
    dst[o + 3] = f2bf(v.w);
  }
}

// ---------------------------------------------------------------------------
// Token-shift mix + bf16 convert:  out = bf16( x + (shift(x) - x) * coef ).
// Rows (B*T) x 1024; batch boundary at m % 2048 == 0 -> shifted row is zero.
// ---------------------------------------------------------------------------
__global__ __launch_bounds__(256)
void mixcvt_kernel(const float* __restrict__ X, const float* __restrict__ coef,
                   __bf16* __restrict__ out)
{
  int i4 = blockIdx.x * 256 + threadIdx.x;  // float4 index (total M*C/4)
  size_t idx = (size_t)i4 * 4;
  int m = (int)(idx >> 10);
  int k = (int)(idx & 1023);
  float4 v = *(const float4*)&X[idx];
  float4 p = make_float4(0.f, 0.f, 0.f, 0.f);
  if (m & 2047) p = *(const float4*)&X[idx - 1024];
  float4 cf = *(const float4*)&coef[k];
  v.x += (p.x - v.x) * cf.x;
  v.y += (p.y - v.y) * cf.y;
  v.z += (p.z - v.z) * cf.z;
  v.w += (p.w - v.w) * cf.w;
  out[idx + 0] = f2bf(v.x);
  out[idx + 1] = f2bf(v.y);
  out[idx + 2] = f2bf(v.z);
  out[idx + 3] = f2bf(v.w);
}

// ---------------------------------------------------------------------------
// Y[m,n] = act( sum_k X[m,k] * W[n,k] + bias[n] )   (torch Linear: x @ W^T)
// X, W bf16; ACT: 0 none, 1 tanh, 2 sigmoid; OUTBF: store bf16 else fp32.
// Block tile 128x128, BK=32. 8 wave32: wave (wvM,wvN) in 2x4 grid owns a
// 64x32 slab = 4 M-frags x 2 N-frags -> 8 v_wmma per K-step.
// ---------------------------------------------------------------------------
#define BM 128
#define BN 128
#define BK 32

template <int ACT, bool OUTBF>
__global__ __launch_bounds__(256)
void gemm_bf16_wmma(const __bf16* __restrict__ X, const __bf16* __restrict__ W,
                    const float* __restrict__ bias, void* __restrict__ Yv,
                    int M, int N, int K)
{
  __shared__ __bf16 As[2][BM][BK + 2];   // [128][34] rows = TDM pad layout
  __shared__ __bf16 Bs[2][BM][BK + 2];

  const int tid   = threadIdx.x;
  const int lane  = tid & 31;
  const int wave  = tid >> 5;
  const int wvM   = wave >> 2;        // 0..1 : 64-row slab
  const int wvN   = wave & 3;         // 0..3 : 32-col slab
  const int mrow  = lane & 15;
  const int khalf = lane >> 4;
  const int m0    = blockIdx.y * BM;
  const int n0    = blockIdx.x * BN;

#if __has_builtin(__builtin_amdgcn_tensor_load_to_lds)
  // ---- TDM issue: 128x32 bf16 tile of `base` (rows x K) into LDS ----
  auto issue_tdm = [&](const __bf16* base, int row0, int RT, int kk,
                       __bf16* lds_dst) {
    unsigned lds_b = (unsigned)(size_t)lds_dst;            // LDS byte offset
    unsigned long long ga =
        (unsigned long long)(const void*)(base + (size_t)row0 * K + kk);
    unsigned rem0 = (unsigned)(K - kk);                    // remaining K extent
    unsigned rem1 = (unsigned)(RT > row0 ? RT - row0 : 0); // remaining rows
    v4u g0;
    g0[0] = 1u;                                            // count=1 valid D#
    g0[1] = lds_b;                                         // lds_addr
    g0[2] = (unsigned)ga;                                  // global_addr[31:0]
    g0[3] = (unsigned)((ga >> 32) & 0x01FFFFFFu) | (2u << 30); // [56:32]|type=2
    v8i g1;
    // data_size=2B (code 1), pad_enable, pad_interval code 3 (=16 DWORDs =
    // one 32-elem bf16 row), pad_amount code 0 (=1 DWORD = 2 bf16)
    //   -> LDS row stride 34 bf16, bank-conflict free.
    g1[0] = (int)((1u << 16) | (1u << 20) | (3u << 22));
    g1[1] = (int)((rem0 & 0xFFFFu) << 16);                 // tensor_dim0 lo
    g1[2] = (int)((rem0 >> 16) | ((rem1 & 0xFFFFu) << 16));
    g1[3] = (int)((rem1 >> 16) | (32u << 16));             // tile_dim0 = 32
    g1[4] = (int)BM;                                       // tile_dim1 = 128
    g1[5] = (int)(unsigned)K;                              // dim0_stride (elems)
    g1[6] = 0;
    g1[7] = 0;
    v4i gz = {0, 0, 0, 0};
#if defined(__clang_major__) && (__clang_major__ >= 23)
    v8i gz8 = {0, 0, 0, 0, 0, 0, 0, 0};
    __builtin_amdgcn_tensor_load_to_lds(g0, g1, gz, gz, gz8, 0);
#else
    __builtin_amdgcn_tensor_load_to_lds(g0, g1, gz, gz, 0);
#endif
  };
  auto stage = [&](int kk, int buf) {
    if (wave == 0) {
      issue_tdm(X, m0, M, kk, &As[buf][0][0]);
      issue_tdm(W, n0, N, kk, &Bs[buf][0][0]);
    }
  };
  auto drain = [&]() {
    if (wave == 0) __builtin_amdgcn_s_wait_tensorcnt((short)0);
  };
#else
  // ---- fallback: manual cooperative staging ----
  auto stage = [&](int kk, int buf) {
    for (int i = tid; i < BM * BK; i += 256) {
      int r = i >> 5, c = i & 31;
      int ma = m0 + r, nb = n0 + r;
      As[buf][r][c] = (ma < M) ? X[(size_t)ma * K + kk + c] : f2bf(0.0f);
      Bs[buf][r][c] = (nb < N) ? W[(size_t)nb * K + kk + c] : f2bf(0.0f);
    }
  };
  auto drain = [&]() {};
#endif

  v8f acc[4][2];
#pragma unroll
  for (int f = 0; f < 4; ++f)
#pragma unroll
    for (int g = 0; g < 2; ++g) acc[f][g] = v8f{};

  // ---- prologue: fill buffer 0 ----
  stage(0, 0);
  drain();
  __syncthreads();

  for (int k0 = 0; k0 < K; k0 += BK) {
    const int cur = (k0 >> 5) & 1;
    const bool more = (k0 + BK) < K;

    // kick the NEXT tile's DMA so it overlaps this tile's WMMAs
    if (more) {
      stage(k0 + BK, cur ^ 1);
      if (k0 + 2 * BK < K) {          // warm L2 two tiles ahead
        int pr = tid >> 1;
        int pc = (tid & 1) << 4;
        int pm = m0 + pr;
        if (pm < M)
          __builtin_prefetch(&X[(size_t)pm * K + k0 + 2 * BK + pc], 0, 0);
      }
    }

    // ---- fragments per ISA 7.12.2 wave32 layouts (pure ds loads) ----
    v16bf afrag[4], bfrag[2];
#pragma unroll
    for (int f = 0; f < 4; ++f)
#pragma unroll
      for (int e = 0; e < 16; ++e) {
        int ka = ((e & 8) << 1) + (khalf << 3) + (e & 7);
        afrag[f][e] = As[cur][(wvM << 6) + (f << 4) + mrow][ka];
      }
#pragma unroll
    for (int g = 0; g < 2; ++g)
#pragma unroll
      for (int e = 0; e < 16; ++e) {
        int kb = ((e & 8) << 1) + (khalf << 3) + (e & 7);
        bfrag[g][e] = Bs[cur][(wvN << 5) + (g << 4) + mrow][kb];
      }
#pragma unroll
    for (int f = 0; f < 4; ++f)
#pragma unroll
      for (int g = 0; g < 2; ++g)
        acc[f][g] = __builtin_amdgcn_wmma_f32_16x16x32_bf16(
            false, afrag[f], false, bfrag[g], (short)0, acc[f][g], false, false);

    // TENSORcnt completes in-order: draining fences both of next tile's DMAs.
    if (more) drain();
    __syncthreads();
  }

  // ---- epilogue: C/D layout M = vgpr + 8*khalf, N = lane&15 ----
#pragma unroll
  for (int f = 0; f < 4; ++f) {
#pragma unroll
    for (int rr = 0; rr < 8; ++rr) {
      int m = m0 + (wvM << 6) + (f << 4) + (khalf << 3) + rr;
      if (m >= M) continue;
#pragma unroll
      for (int g = 0; g < 2; ++g) {
        int n = n0 + (wvN << 5) + (g << 4) + mrow;
        if (n < N) {
          float v = acc[f][g][rr] + (bias ? bias[n] : 0.0f);
          if (ACT == 1) v = tanhf(v);
          if (ACT == 2) v = 1.0f / (1.0f + expf(-v));
          if (OUTBF) ((__bf16*)Yv)[(size_t)m * N + n] = f2bf(v);
          else       ((float*)Yv)[(size_t)m * N + n] = v;
        }
      }
    }
  }
}

// ---------------------------------------------------------------------------
// w0[c] = w0_base + cumsum(softplus(w0_delta))  (serial, 1023 steps, trivial)
// ---------------------------------------------------------------------------
__global__ void w0_kernel(const float* __restrict__ w0_base,
                          const float* __restrict__ w0_delta,
                          float* __restrict__ w0)
{
  if (threadIdx.x == 0 && blockIdx.x == 0) {
    float acc = w0_base[0];
    w0[0] = acc;
    for (int i = 1; i < 1024; ++i) {
      acc += softplus_f(w0_delta[i - 1]);
      w0[i] = acc;
    }
  }
}

// decay = exp(-exp(-softplus(-(w0 + w_lr)) - 0.5)), in place on wbuf
__global__ __launch_bounds__(256)
void decay_kernel(float* __restrict__ wbuf, const float* __restrict__ w0)
{
  int idx = blockIdx.x * 256 + threadIdx.x;
  int c = idx & 1023;
  float w = w0[c] + wbuf[idx];
  float wv = -softplus_f(-w) - 0.5f;
  wbuf[idx] = expf(-expf(wv));
}

// ---------------------------------------------------------------------------
// Per-head prep: kk = normalize(k*k_k), a_in = -kk, b_in = kk*a,
// k <- k*(1+(a-1)*k_a).  4 heads per 256-thread block, 64 lanes per head.
// ---------------------------------------------------------------------------
__global__ __launch_bounds__(256)
void prep_kernel(float* __restrict__ kbuf, const float* __restrict__ abuf,
                 const float* __restrict__ k_k, const float* __restrict__ k_a,
                 float* __restrict__ a_in, float* __restrict__ b_in)
{
  int head = blockIdx.x * 4 + (threadIdx.x >> 6);
  int d = threadIdx.x & 63;
  int bt = head >> 4, h = head & 15;
  int c = (h << 6) + d;
  int idx = (bt << 10) + c;

  float kv = kbuf[idx];
  float av = abuf[idx];
  float kkv = kv * k_k[c];

  __shared__ float red[256];
  int gbase = threadIdx.x & ~63;
  red[threadIdx.x] = kkv * kkv;
  __syncthreads();
  for (int s = 32; s; s >>= 1) {
    if (d < s) red[threadIdx.x] += red[threadIdx.x + s];
    __syncthreads();
  }
  float nrm = sqrtf(red[gbase]);
  float inv = 1.0f / fmaxf(nrm, 1e-12f);
  float kkn = kkv * inv;

  a_in[idx] = -kkn;
  b_in[idx] = kkn * av;
  kbuf[idx] = kv * (1.0f + (av - 1.0f) * k_a[c]);
}

// ---------------------------------------------------------------------------
// RWKV-7 recurrence. 1 block per (b,h); 64 threads; thread i owns state row
// S[i][0..63] in VGPRs. Per-step vectors broadcast through LDS.
//   S = S*diag(w) + (S a) b^T + v k^T ;  o = S r
// ---------------------------------------------------------------------------
__global__ __launch_bounds__(64)
void rwkv7_scan(const float* __restrict__ r, const float* __restrict__ w,
                const float* __restrict__ k, const float* __restrict__ v,
                const float* __restrict__ a_in, const float* __restrict__ b_in,
                float* __restrict__ o)
{
  const int bh = blockIdx.x;           // b*16 + h
  const int b = bh >> 4, h = bh & 15;
  const int i = threadIdx.x;           // value row (0..63)

  float S[64];
#pragma unroll
  for (int j = 0; j < 64; ++j) S[j] = 0.0f;

  __shared__ float rs[64], ws[64], ks[64], as[64], bs[64];

  const size_t base = (size_t)b * 2048u * 1024u + (size_t)(h << 6);
  for (int t = 0; t < 2048; ++t) {
    size_t off = base + (size_t)t * 1024u;
    rs[i] = r[off + i];
    ws[i] = w[off + i];
    ks[i] = k[off + i];
    as[i] = a_in[off + i];
    bs[i] = b_in[off + i];
    float vi = v[off + i];
    __syncthreads();

    float sa = 0.0f;
#pragma unroll
    for (int j = 0; j < 64; ++j) sa += S[j] * as[j];

    float oi = 0.0f;
#pragma unroll
    for (int j = 0; j < 64; ++j) {
      float s = S[j] * ws[j] + sa * bs[j] + vi * ks[j];
      S[j] = s;
      oi += s * rs[j];
    }
    o[off + i] = oi;
    __syncthreads();
  }
}

// ---------------------------------------------------------------------------
// GroupNorm(H groups) + per-head bonus + gate; writes bf16 for the final GEMM:
//   y = bf16( (gn(o) + (r.k.r_k)*v) * g )
// ---------------------------------------------------------------------------
__global__ __launch_bounds__(256)
void post_kernel(const float* __restrict__ o, const float* __restrict__ r,
                 const float* __restrict__ k, const float* __restrict__ v,
                 const float* __restrict__ g, const float* __restrict__ r_k,
                 const float* __restrict__ gnw, const float* __restrict__ gnb,
                 __bf16* __restrict__ y)
{
  int head = blockIdx.x * 4 + (threadIdx.x >> 6);
  int d = threadIdx.x & 63;
  int bt = head >> 4, h = head & 15;
  int c = (h << 6) + d;
  int idx = (bt << 10) + c;

  float ov = o[idx];
  __shared__ float red[256];
  int gbase = threadIdx.x & ~63;

  red[threadIdx.x] = ov;                       // mean
  __syncthreads();
  for (int s = 32; s; s >>= 1) {
    if (d < s) red[threadIdx.x] += red[threadIdx.x + s];
    __syncthreads();
  }
  float mu = red[gbase] * (1.0f / 64.0f);
  __syncthreads();

  red[threadIdx.x] = ov * ov;                  // E[x^2]
  __syncthreads();
  for (int s = 32; s; s >>= 1) {
    if (d < s) red[threadIdx.x] += red[threadIdx.x + s];
    __syncthreads();
  }
  float var = red[gbase] * (1.0f / 64.0f) - mu * mu;
  __syncthreads();

  float rv = r[idx], kv = k[idx];              // bonus dot
  red[threadIdx.x] = rv * kv * r_k[(h << 6) + d];
  __syncthreads();
  for (int s = 32; s; s >>= 1) {
    if (d < s) red[threadIdx.x] += red[threadIdx.x + s];
    __syncthreads();
  }
  float dot = red[gbase];
  __syncthreads();

  const float EPS = 64.0f * 1e-5f;
  float on = (ov - mu) * rsqrtf(var + EPS) * gnw[c] + gnb[c];
  y[idx] = f2bf((on + dot * v[idx]) * g[idx]);
}

// ---------------------------------------------------------------------------
extern "C" void kernel_launch(void* const* d_in, const int* in_sizes, int n_in,
                              void* d_out, int out_size, void* d_ws, size_t ws_size,
                              hipStream_t stream)
{
  (void)in_sizes; (void)n_in; (void)out_size; (void)ws_size;

  const float* hs   = (const float*)d_in[0];
  const float* x_r  = (const float*)d_in[1];
  const float* x_w  = (const float*)d_in[2];
  const float* x_k  = (const float*)d_in[3];
  const float* x_v  = (const float*)d_in[4];
  const float* x_a  = (const float*)d_in[5];
  const float* x_g  = (const float*)d_in[6];
  const float* k_k  = (const float*)d_in[7];
  const float* k_a  = (const float*)d_in[8];
  const float* r_k  = (const float*)d_in[9];
  const float* Wr   = (const float*)d_in[10];
  const float* Wk   = (const float*)d_in[11];
  const float* Wv   = (const float*)d_in[12];
  const float* Wo   = (const float*)d_in[13];
  const float* wA   = (const float*)d_in[14];
  const float* wB   = (const float*)d_in[15];
  const float* wbias= (const float*)d_in[16];
  const float* aA   = (const float*)d_in[17];
  const float* aB   = (const float*)d_in[18];
  const float* abias= (const float*)d_in[19];
  const float* gA   = (const float*)d_in[20];
  const float* gB   = (const float*)d_in[21];
  const float* gnw  = (const float*)d_in[22];
  const float* gnb  = (const float*)d_in[23];
  const float* w0b  = (const float*)d_in[24];
  const float* w0d  = (const float*)d_in[25];

  const int M = 4096, C = 1024, T = 2048;
  (void)T;
  const size_t NE = (size_t)M * C;

  // -------- fp32 workspace --------
  float* fws  = (float*)d_ws;
  float* rbuf = fws;             // r
  float* kbuf = fws + 1 * NE;    // k -> k_adj (in place)
  float* vbuf = fws + 2 * NE;    // v
  float* wbuf = fws + 3 * NE;    // w_lr -> decay (in place)
  float* abuf = fws + 4 * NE;    // a (sigmoid)
  float* gbuf = fws + 5 * NE;    // g
  float* ain  = fws + 6 * NE;    // -kk
  float* bin  = fws + 7 * NE;    // kk*a
  float* obuf = fws + 8 * NE;    // scan output
  float* w0   = fws + 9 * NE;    // 1024

  // -------- bf16 workspace --------
  __bf16* b16  = (__bf16*)(fws + 9 * NE + 1024);
  __bf16* xr16 = b16 + 0 * NE;   // mixed activations
  __bf16* xw16 = b16 + 1 * NE;
  __bf16* xk16 = b16 + 2 * NE;
  __bf16* xv16 = b16 + 3 * NE;
  __bf16* xa16 = b16 + 4 * NE;
  __bf16* xg16 = b16 + 5 * NE;
  __bf16* y16  = b16 + 6 * NE;   // gated output (pre-Wo)
  __bf16* wh16 = b16 + 7 * NE;                  // 4096 x 64
  __bf16* ah16 = wh16 + (size_t)M * 64;         // 4096 x 64
  __bf16* gh16 = ah16 + (size_t)M * 64;         // 4096 x 160
  __bf16* Wr16 = gh16 + (size_t)M * 160;        // weights, bf16 copies
  __bf16* Wk16 = Wr16 + (size_t)C * C;
  __bf16* Wv16 = Wk16 + (size_t)C * C;
  __bf16* Wo16 = Wv16 + (size_t)C * C;
  __bf16* wA16 = Wo16 + (size_t)C * C;          // 64 x C
  __bf16* wB16 = wA16 + (size_t)64 * C;         // C x 64
  __bf16* aA16 = wB16 + (size_t)C * 64;
  __bf16* aB16 = aA16 + (size_t)64 * C;
  __bf16* gA16 = aB16 + (size_t)C * 64;         // 160 x C
  __bf16* gB16 = gA16 + (size_t)160 * C;        // C x 160

  dim3 blk(256);
  auto grd = [](int m, int n) { return dim3((unsigned)((n + BN - 1) / BN),
                                            (unsigned)((m + BM - 1) / BM)); };
  auto cgrd = [](size_t n) { return dim3((unsigned)((n / 4 + 255) / 256)); };

  // -------- one-time bf16 conversions (bandwidth passes) --------
  cvt_kernel<<<cgrd(NE), blk, 0, stream>>>(Wr, Wr16, (int)(NE / 4));
  cvt_kernel<<<cgrd(NE), blk, 0, stream>>>(Wk, Wk16, (int)(NE / 4));
  cvt_kernel<<<cgrd(NE), blk, 0, stream>>>(Wv, Wv16, (int)(NE / 4));
  cvt_kernel<<<cgrd(NE), blk, 0, stream>>>(Wo, Wo16, (int)(NE / 4));
  cvt_kernel<<<cgrd(64 * C), blk, 0, stream>>>(wA, wA16, 64 * C / 4);
  cvt_kernel<<<cgrd(C * 64), blk, 0, stream>>>(wB, wB16, C * 64 / 4);
  cvt_kernel<<<cgrd(64 * C), blk, 0, stream>>>(aA, aA16, 64 * C / 4);
  cvt_kernel<<<cgrd(C * 64), blk, 0, stream>>>(aB, aB16, C * 64 / 4);
  cvt_kernel<<<cgrd(160 * C), blk, 0, stream>>>(gA, gA16, 160 * C / 4);
  cvt_kernel<<<cgrd(C * 160), blk, 0, stream>>>(gB, gB16, C * 160 / 4);

  // token-shift mixes, materialized bf16
  mixcvt_kernel<<<(unsigned)(NE / 1024), blk, 0, stream>>>(hs, x_r, xr16);
  mixcvt_kernel<<<(unsigned)(NE / 1024), blk, 0, stream>>>(hs, x_w, xw16);
  mixcvt_kernel<<<(unsigned)(NE / 1024), blk, 0, stream>>>(hs, x_k, xk16);
  mixcvt_kernel<<<(unsigned)(NE / 1024), blk, 0, stream>>>(hs, x_v, xv16);
  mixcvt_kernel<<<(unsigned)(NE / 1024), blk, 0, stream>>>(hs, x_a, xa16);
  mixcvt_kernel<<<(unsigned)(NE / 1024), blk, 0, stream>>>(hs, x_g, xg16);

  // -------- projections (all-bf16 WMMA GEMMs, TDM-staged tiles) --------
  gemm_bf16_wmma<0, false><<<grd(M, C), blk, 0, stream>>>(xr16, Wr16, nullptr, rbuf, M, C, C);
  gemm_bf16_wmma<0, false><<<grd(M, C), blk, 0, stream>>>(xk16, Wk16, nullptr, kbuf, M, C, C);
  gemm_bf16_wmma<0, false><<<grd(M, C), blk, 0, stream>>>(xv16, Wv16, nullptr, vbuf, M, C, C);

  // decay LoRA: tanh(xw @ wA^T) @ wB^T + bias
  gemm_bf16_wmma<1, true ><<<grd(M, 64), blk, 0, stream>>>(xw16, wA16, nullptr, wh16, M, 64, C);
  gemm_bf16_wmma<0, false><<<grd(M, C), blk, 0, stream>>>(wh16, wB16, wbias, wbuf, M, C, 64);

  // icl-rate LoRA: sigmoid((xa @ aA^T) @ aB^T + bias)
  gemm_bf16_wmma<0, true ><<<grd(M, 64), blk, 0, stream>>>(xa16, aA16, nullptr, ah16, M, 64, C);
  gemm_bf16_wmma<2, false><<<grd(M, C), blk, 0, stream>>>(ah16, aB16, abias, abuf, M, C, 64);

  // gate LoRA: sigmoid(xg @ gA^T) @ gB^T
  gemm_bf16_wmma<2, true ><<<grd(M, 160), blk, 0, stream>>>(xg16, gA16, nullptr, gh16, M, 160, C);
  gemm_bf16_wmma<0, false><<<grd(M, C), blk, 0, stream>>>(gh16, gB16, nullptr, gbuf, M, C, 160);

  // -------- PoST decay parameterization --------
  w0_kernel<<<1, 64, 0, stream>>>(w0b, w0d, w0);
  decay_kernel<<<(unsigned)(NE / 256), blk, 0, stream>>>(wbuf, w0);

  // kk normalize / a_in / b_in / k adjust
  prep_kernel<<<16384, blk, 0, stream>>>(kbuf, abuf, k_k, k_a, ain, bin);

  // sequential RWKV-7 state scan: 32 chains (B*H), state in VGPRs (fp32)
  rwkv7_scan<<<32, 64, 0, stream>>>(rbuf, wbuf, kbuf, vbuf, ain, bin, obuf);

  // GroupNorm + bonus + gate -> bf16
  post_kernel<<<16384, blk, 0, stream>>>(obuf, rbuf, kbuf, vbuf, gbuf, r_k, gnw, gnb, y16);

  // output projection (fp32 out)
  gemm_bf16_wmma<0, false><<<grd(M, C), blk, 0, stream>>>(y16, Wo16, nullptr,
                                                          (float*)d_out, M, C, C);
}
